// ReferenceImpl_36696200577057
// MI455X (gfx1250) — matrix operand order repaired and checked by reading, not compiled, not run on hardware
//
#include <hip/hip_runtime.h>

#define G_GRP   8
#define DIN     1024
#define DOUT    1024
#define M_TOT   8192
#define BM      128
#define BN      128
#define BK      32
#define KSTEPS  (DIN / BK)   // 32
#define LDR     40           // padded LDS row length (bf16 elems): 80B stride

typedef __attribute__((ext_vector_type(16))) __bf16 v16bf;
typedef __attribute__((ext_vector_type(8)))  float  v8f;

struct FragB32 { uint4 a, b; };  // 32 bytes = 16 bf16 (one wmma operand per lane)

__device__ __forceinline__ void cvt_store4(unsigned short* hp, unsigned short* lp, float4 v) {
    // split 4 f32 into hi-bf16 (truncate) and lo-bf16 (residual), packed 2-per-dword
    unsigned b0 = __float_as_uint(v.x), b1 = __float_as_uint(v.y);
    unsigned b2 = __float_as_uint(v.z), b3 = __float_as_uint(v.w);
    unsigned h0 = b0 & 0xFFFF0000u, h1 = b1 & 0xFFFF0000u;
    unsigned h2 = b2 & 0xFFFF0000u, h3 = b3 & 0xFFFF0000u;
    float l0 = v.x - __uint_as_float(h0);
    float l1 = v.y - __uint_as_float(h1);
    float l2 = v.z - __uint_as_float(h2);
    float l3 = v.w - __uint_as_float(h3);
    ((unsigned*)hp)[0] = h1 | (h0 >> 16);
    ((unsigned*)hp)[1] = h3 | (h2 >> 16);
    ((unsigned*)lp)[0] = (__float_as_uint(l1) & 0xFFFF0000u) | (__float_as_uint(l0) >> 16);
    ((unsigned*)lp)[1] = (__float_as_uint(l3) & 0xFFFF0000u) | (__float_as_uint(l2) >> 16);
}

__global__ __launch_bounds__(256) void grouped_gemm_bf16x3(
    const float* __restrict__ X, const float* __restrict__ W, float* __restrict__ out)
{
#if defined(__gfx1250__)
    // planes: 0 = A_hi, 1 = A_lo, 2 = B_hi, 3 = B_lo ; [2] = double buffer (80 KB)
    __shared__ __align__(16) unsigned short smem[2][4][BM][LDR];

    const int g   = blockIdx.z;
    const int mb  = blockIdx.y * BM;
    const int nb  = blockIdx.x * BN;
    const int tid = threadIdx.x;
    const int lane = tid & 31;          // wave32
    const int wave = tid >> 5;          // 0..7
    const int wm   = wave & 1;          // 2 waves along M (64 rows each)
    const int wn   = wave >> 1;         // 4 waves along N (32 cols each)
    const int lr   = lane & 15;
    const int lhi  = lane >> 4;         // half-wave selector (K-phase)

    // staging assignment: thread -> (row, 16-float half-row) of each 128x32 tile
    const int srow = tid >> 1;          // 0..127
    const int scol = (tid & 1) * 16;    // 0 or 16 (floats)
    const float* gA = X + (((size_t)(mb + srow) * G_GRP + g) * DIN) + scol;
    const float* gB = W + ((size_t)g * DOUT * DIN) + ((size_t)(nb + srow) * DIN) + scol;

    v8f acc[4][2];
    {
        v8f zero = {};
        #pragma unroll
        for (int i = 0; i < 4; ++i)
            #pragma unroll
            for (int j = 0; j < 2; ++j) acc[i][j] = zero;
    }

    float4 ra[4], rb[4];
    #pragma unroll
    for (int t = 0; t < 4; ++t) {
        ra[t] = ((const float4*)gA)[t];
        rb[t] = ((const float4*)gB)[t];
    }
    {   // prologue: fill buffer 0
        unsigned short* ah = &smem[0][0][srow][scol];
        unsigned short* al = &smem[0][1][srow][scol];
        unsigned short* bh = &smem[0][2][srow][scol];
        unsigned short* bl = &smem[0][3][srow][scol];
        #pragma unroll
        for (int t = 0; t < 4; ++t) {
            cvt_store4(ah + 4*t, al + 4*t, ra[t]);
            cvt_store4(bh + 4*t, bl + 4*t, rb[t]);
        }
    }
    __syncthreads();

    for (int step = 0; step < KSTEPS; ++step) {
        const int buf = step & 1;

        if (step + 1 < KSTEPS) {    // register prefetch of next K-slab
            const float* pA = gA + (step + 1) * BK;
            const float* pB = gB + (step + 1) * BK;
            #pragma unroll
            for (int t = 0; t < 4; ++t) {
                ra[t] = ((const float4*)pA)[t];
                rb[t] = ((const float4*)pB)[t];
            }
        }

        // B fragments (32x16 bf16: lanes<16 hold K0..15 contiguously)
        FragB32 bHi[2], bLo[2];
        #pragma unroll
        for (int j = 0; j < 2; ++j) {
            const int brow = wn * 32 + j * 16 + lr;
            const unsigned short* rh = &smem[buf][2][brow][lhi * 16];
            const unsigned short* rl = &smem[buf][3][brow][lhi * 16];
            bHi[j].a = ((const uint4*)rh)[0]; bHi[j].b = ((const uint4*)rh)[1];
            bLo[j].a = ((const uint4*)rl)[0]; bLo[j].b = ((const uint4*)rl)[1];
        }

        // A fragments + 3-term bf16 WMMA (hi*hi + hi*lo + lo*hi)
        #pragma unroll
        for (int i = 0; i < 4; ++i) {
            const int arow = wm * 64 + i * 16 + lr;
            const unsigned short* rh = &smem[buf][0][arow][0];
            const unsigned short* rl = &smem[buf][1][arow][0];
            FragB32 fh, fl;
            fh.a = *(const uint4*)(rh + lhi * 16);
            fh.b = *(const uint4*)(rh + lhi * 16 + 8);
            fl.a = *(const uint4*)(rl + lhi * 16);
            fl.b = *(const uint4*)(rl + lhi * 16 + 8);
            v16bf ah = __builtin_bit_cast(v16bf, fh);
            v16bf al = __builtin_bit_cast(v16bf, fl);
            #pragma unroll
            for (int j = 0; j < 2; ++j) {
                v16bf bh = __builtin_bit_cast(v16bf, bHi[j]);
                v16bf bl = __builtin_bit_cast(v16bf, bLo[j]);
                acc[i][j] = __builtin_amdgcn_wmma_f32_16x16x32_bf16(
                    false, ah, false, bh, (short)0, acc[i][j], false, false);
                acc[i][j] = __builtin_amdgcn_wmma_f32_16x16x32_bf16(
                    false, ah, false, bl, (short)0, acc[i][j], false, false);
                acc[i][j] = __builtin_amdgcn_wmma_f32_16x16x32_bf16(
                    false, al, false, bh, (short)0, acc[i][j], false, false);
            }
        }

        __syncthreads();   // all waves done reading smem[buf]

        if (step + 1 < KSTEPS) {   // fill the other buffer
            const int nbuf = buf ^ 1;
            unsigned short* ah = &smem[nbuf][0][srow][scol];
            unsigned short* al = &smem[nbuf][1][srow][scol];
            unsigned short* bh = &smem[nbuf][2][srow][scol];
            unsigned short* bl = &smem[nbuf][3][srow][scol];
            #pragma unroll
            for (int t = 0; t < 4; ++t) {
                cvt_store4(ah + 4*t, al + 4*t, ra[t]);
                cvt_store4(bh + 4*t, bl + 4*t, rb[t]);
            }
        }
        __syncthreads();
    }

    // write back: C/D layout: VGPR r -> M = r (+8 for lanes>=16), N = lane%16
    #pragma unroll
    for (int i = 0; i < 4; ++i) {
        const int m0 = mb + wm * 64 + i * 16 + lhi * 8;
        #pragma unroll
        for (int j = 0; j < 2; ++j) {
            const int n = nb + wn * 32 + j * 16 + lr;
            float* o = out + ((size_t)m0 * G_GRP + g) * DOUT + n;
            #pragma unroll
            for (int r = 0; r < 8; ++r) {
                o[(size_t)r * G_GRP * DOUT] = acc[i][j][r];
            }
        }
    }
#else
    (void)X; (void)W; (void)out;
#endif
}

extern "C" void kernel_launch(void* const* d_in, const int* in_sizes, int n_in,
                              void* d_out, int out_size, void* d_ws, size_t ws_size,
                              hipStream_t stream) {
    (void)in_sizes; (void)n_in; (void)out_size; (void)d_ws; (void)ws_size;
    const float* x = (const float*)d_in[0];   // [65536, 1024] f32
    const float* W = (const float*)d_in[1];   // [8, 1024, 1024] f32, [g][d][k]
    float* out = (float*)d_out;               // [65536, 1024] f32

    dim3 grid(DOUT / BN, M_TOT / BM, G_GRP);  // (8, 64, 8)
    dim3 block(256, 1, 1);
    grouped_gemm_bf16x3<<<grid, block, 0, stream>>>(x, W, out);
}
